// MHA_27109833572959
// MI455X (gfx1250) — compile-verified, matrix operand order checked by reading, not run on hardware
//
#include <hip/hip_runtime.h>
#include <hip/hip_bf16.h>

// ---------------------------------------------------------------------------
// MHA forward for gfx1250 (MI455X): f16 WMMA (f32 accum) + TDM-staged K/V.
// d_out = [ out (2*2048*1024 f32) | attn_weights (2*16*2048*2048 f32) ]
// ---------------------------------------------------------------------------

typedef _Float16     h16   __attribute__((ext_vector_type(16)));
typedef _Float16     h8    __attribute__((ext_vector_type(8)));
typedef float        f8    __attribute__((ext_vector_type(8)));
typedef unsigned int u32x4 __attribute__((ext_vector_type(4)));
typedef int          i32x4 __attribute__((ext_vector_type(4)));
typedef int          i32x8 __attribute__((ext_vector_type(8)));

#define DMODEL   1024
#define NHEAD    16
#define DK       64
#define SEQ      2048
#define BATCH    2
#define ROWS     (BATCH * SEQ)          // 4096

// Workspace layout (in _Float16 elements)
#define WS_XH    ((size_t)0)                       // X in f16            [4096][1024]
#define WS_WQT   ((size_t)4194304)                 // Wq^T f16            [n][k]
#define WS_WKT   ((size_t)5242880)
#define WS_WVT   ((size_t)6291456)
#define WS_WOT   ((size_t)7340032)
#define WS_QH    ((size_t)8388608)                 // Q  [B][H][S][dk]
#define WS_KH    ((size_t)12582912)                // K  [B][H][S][dk]
#define WS_VT    ((size_t)16777216)                // V  [B][H][dk][S]
#define WS_CTX   ((size_t)20971520)                // attention context [4096][1024]

// ---------------------------------------------------------------------------
// Fragment loader: 16-bit A/B WMMA fragment from a row-major tile.
// Per ISA layout: lane L<16 -> row L, K halves {0..7, 16..23};
//                 lane L>=16 -> row L-16, K halves {8..15, 24..31}.
// P points at tile origin, ld = leading dimension in halves. 16B aligned.
// Works for both global and LDS source pointers (compiler picks b128 op).
// ---------------------------------------------------------------------------
__device__ __forceinline__ h16 frag_ld32(const _Float16* P, int ld) {
    int lane = threadIdx.x & 31;
    int r    = lane & 15;
    int kb   = (lane >> 4) * 8;               // 0 or 8
    const h8* p0 = (const h8*)(P + (size_t)r * ld + kb);
    const h8* p1 = (const h8*)(P + (size_t)r * ld + kb + 16);
    h8 lo = *p0;
    h8 hi = *p1;
    h16 a;
#pragma unroll
    for (int i = 0; i < 8; ++i) { a[i] = lo[i]; a[i + 8] = hi[i]; }
    return a;
}

__device__ __forceinline__ f8 wmma16(h16 a, h16 b, f8 c) {
    return __builtin_amdgcn_wmma_f32_16x16x32_f16(false, a, false, b,
                                                  (short)0, c, false, false);
}

// ---------------------------------------------------------------------------
// Tensor Data Mover: DMA a 2-D f16 tile (tile_d0 contiguous elems per row,
// tile_d1 rows, row stride stride0 elems) from global into LDS (packed
// row-major). Descriptor per CDNA5 ISA ch. 8. Tiles are always fully
// in-bounds here, so tensor dims == tile dims (no OOB handling needed).
// Issue from ONE wave only; completion via s_wait_tensorcnt.
// ---------------------------------------------------------------------------
__device__ __forceinline__ void tdm_load_tile_f16(
    const _Float16* gsrc, _Float16* ldst,
    unsigned tile_d0, unsigned tile_d1, unsigned stride0_elems)
{
    unsigned long long ga = (unsigned long long)(uintptr_t)gsrc;
    unsigned la = (unsigned)(uintptr_t)ldst;   // low 32 bits of generic = LDS byte addr

    u32x4 g0;
    g0.x = 1u;                                              // count=1 (valid), no gather
    g0.y = la;                                              // lds_addr (bytes)
    g0.z = (unsigned)ga;                                    // global_addr[31:0]
    g0.w = (unsigned)((ga >> 32) & 0x01FFFFFFu) | (2u << 30); // addr[56:32] | type=2

    i32x8 g1;
    g1[0] = (int)(1u << 16);                  // data_size=1 (2B); no multicast/pad/iter
    g1[1] = (int)((tile_d0 & 0xFFFFu) << 16); // tensor_dim0[15:0] (hi bits land in g1[2])
    g1[2] = (int)(((tile_d0 >> 16) & 0xFFFFu) | ((tile_d1 & 0xFFFFu) << 16)); // td0 hi | td1 lo
    g1[3] = (int)(((tile_d1 >> 16) & 0xFFFFu) | ((tile_d0 & 0xFFFFu) << 16)); // td1 hi | tile_dim0
    g1[4] = (int)(tile_d1 & 0xFFFFu);         // tile_dim1 (tile_dim2 = 0 -> unused)
    g1[5] = (int)stride0_elems;               // tensor_dim0_stride[31:0]
    g1[6] = 0;                                // stride0[47:32]=0 | dim1_stride lo (unused)
    g1[7] = 0;

    i32x4 z4 = (i32x4){0, 0, 0, 0};
#if defined(__clang_major__) && (__clang_major__ >= 23)
    i32x8 z8 = (i32x8){0, 0, 0, 0, 0, 0, 0, 0};
    __builtin_amdgcn_tensor_load_to_lds(g0, g1, z4, z4, z8, 0);
#else
    __builtin_amdgcn_tensor_load_to_lds(g0, g1, z4, z4, 0);
#endif
}

// ---------------------------------------------------------------------------
// fp32 -> f16 elementwise convert
// ---------------------------------------------------------------------------
__global__ void mha_cvt_f16(const float* __restrict__ src,
                            _Float16* __restrict__ dst, int n) {
    int i = blockIdx.x * 256 + threadIdx.x;
    if (i < n) dst[i] = (_Float16)src[i];
}

// fp32 W[k][n] -> f16 Wt[n][k]
__global__ void mha_cvt_transpose(const float* __restrict__ W,
                                  _Float16* __restrict__ Wt) {
    int i = blockIdx.x * 256 + threadIdx.x;     // 1M elements
    int k = i >> 10, n = i & 1023;
    Wt[(size_t)n * DMODEL + k] = (_Float16)W[i];
}

// ---------------------------------------------------------------------------
// Projection GEMM: C[m,n] = A[m,:] @ W[:,n] + bias[n]; W given transposed.
// Block = 128 threads (4 waves). Wave tile = 32x64. Block tile = 32x256.
// mode 0: f16 out, scatter [B][H][S][dk]     (Q, K)
// mode 1: f16 out, scatter [B][H][dk][S]     (V, pre-transposed for AV)
// mode 2: f32 out, plain [m][n]              (final projection)
// ---------------------------------------------------------------------------
__global__ __launch_bounds__(128) void mha_gemm_proj(
    const _Float16* __restrict__ A,     // [4096][1024]
    const _Float16* __restrict__ Bt,    // [1024(n)][1024(k)]
    const float*    __restrict__ bias,  // [1024]
    _Float16* __restrict__ outH,
    float*    __restrict__ outF,
    int mode)
{
    int wave = threadIdx.x >> 5;
    int lane = threadIdx.x & 31;
    int m0 = blockIdx.x * 32;
    int n0 = blockIdx.y * 256 + wave * 64;

    f8 acc[2][4];
#pragma unroll
    for (int i = 0; i < 2; ++i)
#pragma unroll
        for (int j = 0; j < 4; ++j) acc[i][j] = (f8){0,0,0,0,0,0,0,0};

    for (int k = 0; k < DMODEL; k += 32) {
        h16 a0 = frag_ld32(A + (size_t)(m0     ) * DMODEL + k, DMODEL);
        h16 a1 = frag_ld32(A + (size_t)(m0 + 16) * DMODEL + k, DMODEL);
#pragma unroll
        for (int nf = 0; nf < 4; ++nf) {
            h16 b = frag_ld32(Bt + (size_t)(n0 + nf * 16) * DMODEL + k, DMODEL);
            acc[0][nf] = wmma16(a0, b, acc[0][nf]);
            acc[1][nf] = wmma16(a1, b, acc[1][nf]);
        }
    }

    // Store with bias. C layout: lane<16 -> N=lane, M=v; lane>=16 -> N=lane-16, M=v+8
#pragma unroll
    for (int nf = 0; nf < 4; ++nf) {
        int n  = n0 + nf * 16 + (lane & 15);
        float bv = bias[n];
        int hh = n >> 6, dd = n & 63;
#pragma unroll
        for (int mf = 0; mf < 2; ++mf) {
#pragma unroll
            for (int r = 0; r < 8; ++r) {
                int row = m0 + mf * 16 + ((lane < 16) ? r : r + 8);
                float v = acc[mf][nf][r] + bv;
                if (mode == 2) {
                    outF[(size_t)row * DMODEL + n] = v;
                } else {
                    int bb = row >> 11, ss = row & (SEQ - 1);
                    size_t idx;
                    if (mode == 0)  // [B][H][S][dk]
                        idx = ((((size_t)bb * NHEAD + hh) * SEQ + ss) * DK) + dd;
                    else            // [B][H][dk][S]
                        idx = ((((size_t)bb * NHEAD + hh) * DK + dd) * SEQ) + ss;
                    outH[idx] = (_Float16)v;
                }
            }
        }
    }
}

// ---------------------------------------------------------------------------
// Fused attention. 4 waves/WG, 16 queries/wave, key blocks of 32.
// K (32x64) and V (64x32) tiles are DMA'd into LDS by the TDM (wave 0),
// double-buffered so tile i+1 streams in while tile i is consumed; all
// 4 waves share the staged tiles (4x less L2 traffic than direct loads).
// Pass 1: online softmax stats via WMMA QK^T + shfl_xor reductions.
// Pass 2: recompute scores, p = exp(s-m)/l, bounce p through LDS for
//         coalesced one-shot attn_weights stores + A-fragment AV WMMA.
// ---------------------------------------------------------------------------
__global__ __launch_bounds__(128) void mha_attn(
    const _Float16* __restrict__ Qh,   // [B][H][S][dk]
    const _Float16* __restrict__ Kh,   // [B][H][S][dk]
    const _Float16* __restrict__ Vt,   // [B][H][dk][S]
    float*    __restrict__ attnW,      // [B][H][S][S]
    _Float16* __restrict__ Ctx)        // [B][S][H*dk]
{
    __shared__ __attribute__((aligned(16))) _Float16 Kl[2][32 * DK];   // 8 KB
    __shared__ __attribute__((aligned(16))) _Float16 Vl[2][DK * 32];   // 8 KB
    __shared__ __attribute__((aligned(16))) float    ldsP[4][16 * 32]; // 8 KB

    int wave = threadIdx.x >> 5;
    int lane = threadIdx.x & 31;
    int h = blockIdx.y, b = blockIdx.z;
    int q0 = blockIdx.x * 64 + wave * 16;

    const _Float16* Qb = Qh + (((size_t)b * NHEAD + h) * SEQ) * DK;
    const _Float16* Kb = Kh + (((size_t)b * NHEAD + h) * SEQ) * DK;
    const _Float16* Vb = Vt + (((size_t)b * NHEAD + h) * DK) * SEQ;

    const float scale = 0.125f;          // 1/sqrt(64)

    h16 qlo = frag_ld32(Qb + (size_t)q0 * DK,      DK);   // dk 0..31
    h16 qhi = frag_ld32(Qb + (size_t)q0 * DK + 32, DK);   // dk 32..63

    float mrun[8], lrun[8];
#pragma unroll
    for (int r = 0; r < 8; ++r) { mrun[r] = -1e30f; lrun[r] = 0.f; }

    // ---------------- pass 1: softmax statistics (K via TDM) ----------------
    if (wave == 0) tdm_load_tile_f16(Kb, Kl[0], DK, 32, DK);
    int buf = 0;
    for (int kb = 0; kb < SEQ; kb += 32) {
        if (wave == 0) __builtin_amdgcn_s_wait_tensorcnt(0);
        __syncthreads();                               // tile[buf] visible to all
        if (wave == 0 && kb + 32 < SEQ)
            tdm_load_tile_f16(Kb + (size_t)(kb + 32) * DK, Kl[buf ^ 1], DK, 32, DK);

        const _Float16* Kc = Kl[buf];
        h16 b00 = frag_ld32(Kc,            DK);
        h16 b01 = frag_ld32(Kc + 32,       DK);
        h16 b10 = frag_ld32(Kc + 16 * DK,      DK);
        h16 b11 = frag_ld32(Kc + 16 * DK + 32, DK);
        f8 s0 = (f8){0,0,0,0,0,0,0,0}, s1 = (f8){0,0,0,0,0,0,0,0};
        s0 = wmma16(qlo, b00, s0); s0 = wmma16(qhi, b01, s0);
        s1 = wmma16(qlo, b10, s1); s1 = wmma16(qhi, b11, s1);
#pragma unroll
        for (int r = 0; r < 8; ++r) {
            float tm = fmaxf(s0[r], s1[r]);
            tm = fmaxf(tm, __shfl_xor(tm, 1));
            tm = fmaxf(tm, __shfl_xor(tm, 2));
            tm = fmaxf(tm, __shfl_xor(tm, 4));
            tm = fmaxf(tm, __shfl_xor(tm, 8));
            float mn = fmaxf(mrun[r], tm * scale);
            float e = __expf(s0[r] * scale - mn) + __expf(s1[r] * scale - mn);
            e += __shfl_xor(e, 1);
            e += __shfl_xor(e, 2);
            e += __shfl_xor(e, 4);
            e += __shfl_xor(e, 8);
            lrun[r] = lrun[r] * __expf(mrun[r] - mn) + e;
            mrun[r] = mn;
        }
        __syncthreads();                               // done reading tile[buf]
        buf ^= 1;
    }
    float inv[8];
#pragma unroll
    for (int r = 0; r < 8; ++r) inv[r] = 1.0f / lrun[r];

    // ---------------- pass 2: p, attn store, AV (K+V via TDM) ----------------
    f8 o[4];
#pragma unroll
    for (int nb = 0; nb < 4; ++nb) o[nb] = (f8){0,0,0,0,0,0,0,0};

    float* myP = ldsP[wave];
    float* abase = attnW + ((((size_t)b * NHEAD + h) * SEQ + q0) * SEQ);

    if (wave == 0) {
        tdm_load_tile_f16(Kb, Kl[0], DK, 32, DK);
        tdm_load_tile_f16(Vb, Vl[0], 32, DK, SEQ);
    }
    buf = 0;
    for (int kb = 0; kb < SEQ; kb += 32) {
        if (wave == 0) __builtin_amdgcn_s_wait_tensorcnt(0);
        __syncthreads();
        if (wave == 0 && kb + 32 < SEQ) {
            tdm_load_tile_f16(Kb + (size_t)(kb + 32) * DK, Kl[buf ^ 1], DK, 32, DK);
            tdm_load_tile_f16(Vb + (kb + 32),              Vl[buf ^ 1], 32, DK, SEQ);
        }

        const _Float16* Kc = Kl[buf];
        h16 b00 = frag_ld32(Kc,            DK);
        h16 b01 = frag_ld32(Kc + 32,       DK);
        h16 b10 = frag_ld32(Kc + 16 * DK,      DK);
        h16 b11 = frag_ld32(Kc + 16 * DK + 32, DK);
        f8 s0 = (f8){0,0,0,0,0,0,0,0}, s1 = (f8){0,0,0,0,0,0,0,0};
        s0 = wmma16(qlo, b00, s0); s0 = wmma16(qhi, b01, s0);
        s1 = wmma16(qlo, b10, s1); s1 = wmma16(qhi, b11, s1);

#pragma unroll
        for (int r = 0; r < 8; ++r) {
            float p0 = __expf(s0[r] * scale - mrun[r]) * inv[r];
            float p1 = __expf(s1[r] * scale - mrun[r]) * inv[r];
            int mr = (lane < 16) ? r : r + 8;
            int c  = lane & 15;
            myP[mr * 32 + c]      = p0;
            myP[mr * 32 + c + 16] = p1;
        }
        __syncthreads();

        // A-fragment of p (f16) from LDS; AV with LDS-staged V
        {
            int rr = lane & 15, cb = (lane >> 4) * 8;
            h16 pa;
#pragma unroll
            for (int i = 0; i < 8; ++i) {
                pa[i]     = (_Float16)myP[rr * 32 + cb + i];
                pa[i + 8] = (_Float16)myP[rr * 32 + cb + 16 + i];
            }
            const _Float16* Vc = Vl[buf];
#pragma unroll
            for (int nb = 0; nb < 4; ++nb) {
                h16 bv = frag_ld32(Vc + (size_t)(nb * 16) * 32, 32);
                o[nb] = wmma16(pa, bv, o[nb]);
            }
        }

        // Coalesced attn store: 16 rows x 32 contiguous floats (128B/row)
#pragma unroll
        for (int i = 0; i < 16; ++i)
            abase[(size_t)i * SEQ + kb + lane] = myP[i * 32 + lane];

        __syncthreads();                               // p tile + K/V tile consumed
        buf ^= 1;
    }

    // Store context [B][S][H*dk]
#pragma unroll
    for (int nb = 0; nb < 4; ++nb) {
        int d = h * DK + nb * 16 + (lane & 15);
#pragma unroll
        for (int r = 0; r < 8; ++r) {
            int mr = (lane < 16) ? r : r + 8;
            Ctx[(((size_t)b * SEQ + q0 + mr) * DMODEL) + d] = (_Float16)o[nb][r];
        }
    }
}

// ---------------------------------------------------------------------------
extern "C" void kernel_launch(void* const* d_in, const int* in_sizes, int n_in,
                              void* d_out, int out_size, void* d_ws, size_t ws_size,
                              hipStream_t stream) {
    const float* x  = (const float*)d_in[0];
    const float* Wq = (const float*)d_in[1];
    const float* bq = (const float*)d_in[2];
    const float* Wk = (const float*)d_in[3];
    const float* bk = (const float*)d_in[4];
    const float* Wv = (const float*)d_in[5];
    const float* bv = (const float*)d_in[6];
    const float* Wo = (const float*)d_in[7];
    const float* bo = (const float*)d_in[8];

    _Float16* ws  = (_Float16*)d_ws;
    _Float16* Xh  = ws + WS_XH;
    _Float16* Wqt = ws + WS_WQT;
    _Float16* Wkt = ws + WS_WKT;
    _Float16* Wvt = ws + WS_WVT;
    _Float16* Wot = ws + WS_WOT;
    _Float16* Qh  = ws + WS_QH;
    _Float16* Kh  = ws + WS_KH;
    _Float16* Vt  = ws + WS_VT;
    _Float16* Ctx = ws + WS_CTX;

    float* out   = (float*)d_out;
    float* attnW = out + (size_t)ROWS * DMODEL;

    const int NX = ROWS * DMODEL;          // 4,194,304
    mha_cvt_f16<<<NX / 256, 256, 0, stream>>>(x, Xh, NX);
    mha_cvt_transpose<<<(DMODEL * DMODEL) / 256, 256, 0, stream>>>(Wq, Wqt);
    mha_cvt_transpose<<<(DMODEL * DMODEL) / 256, 256, 0, stream>>>(Wk, Wkt);
    mha_cvt_transpose<<<(DMODEL * DMODEL) / 256, 256, 0, stream>>>(Wv, Wvt);
    mha_cvt_transpose<<<(DMODEL * DMODEL) / 256, 256, 0, stream>>>(Wo, Wot);

    dim3 gg(ROWS / 32, DMODEL / 256), bb(128);
    mha_gemm_proj<<<gg, bb, 0, stream>>>(Xh, Wqt, bq, Qh, nullptr, 0);
    mha_gemm_proj<<<gg, bb, 0, stream>>>(Xh, Wkt, bk, Kh, nullptr, 0);
    mha_gemm_proj<<<gg, bb, 0, stream>>>(Xh, Wvt, bv, Vt, nullptr, 1);

    mha_attn<<<dim3(SEQ / 64, NHEAD, BATCH), 128, 0, stream>>>(Qh, Kh, Vt, attnW, Ctx);

    mha_gemm_proj<<<gg, bb, 0, stream>>>(Ctx, Wot, bo, nullptr, out, 2);
}
